// Model_20710332301734
// MI455X (gfx1250) — matrix operand order repaired and checked by reading, not compiled, not run on hardware
//
#include <hip/hip_runtime.h>
#include <hip/hip_bf16.h>
#include <cstdint>

typedef __bf16 bf16;
typedef __attribute__((ext_vector_type(16))) __bf16 v16bf;
typedef __attribute__((ext_vector_type(8)))  __bf16 v8bf;
typedef __attribute__((ext_vector_type(8)))  float v8f;

#define BN_EPS 1e-5f

__device__ __forceinline__ bf16 f2bf(float x) {
  unsigned u; __builtin_memcpy(&u, &x, 4);
  unsigned r = u + 0x7FFFu + ((u >> 16) & 1u);
  unsigned short h = (unsigned short)(r >> 16);
  bf16 o; __builtin_memcpy(&o, &h, 2); return o;
}
__device__ __forceinline__ bf16 bf0() {
  unsigned short h = 0; bf16 o; __builtin_memcpy(&o, &h, 2); return o;
}

// A-fragment: lane already offset by half*16; needs K rel {0..15}.
__device__ __forceinline__ v16bf ld_afrag(const bf16* __restrict__ p) {
  v8bf x = *(const v8bf*)p;          // global_load_b128
  v8bf y = *(const v8bf*)(p + 8);    // global_load_b128
  v16bf r;
#pragma unroll
  for (int e = 0; e < 8; ++e) { r[e] = x[e]; r[8 + e] = y[e]; }
  return r;
}
// B-fragment: lane already offset by half*16; needs K rel {0..15}.
__device__ __forceinline__ v16bf ld_bfrag(const bf16* __restrict__ p) {
  v8bf x = *(const v8bf*)p;          // global_load_b128
  v8bf y = *(const v8bf*)(p + 8);    // global_load_b128
  v16bf r;
#pragma unroll
  for (int e = 0; e < 8; ++e) { r[e] = x[e]; r[8 + e] = y[e]; }
  return r;
}

// ---------------------------------------------------------------------------
// WMMA GEMM: C[r,n] = sum_k A[r,k]*Bt[n,k] (+bias[n]) (optional relu)
// A: bf16 [R,K] row-major. Bt: bf16 [N,K] (B column n contiguous).
// R % 64 == 0, N % 64 == 0. K arbitrary (guarded tail for K % 32 != 0).
// Block = 4 waves; wave -> 16x64 strip (1 A fragment reused for 4 B tiles).
// Grid = (R/64, N/64).
// ---------------------------------------------------------------------------
__global__ void __launch_bounds__(128)
k_gemm_bf16(const bf16* __restrict__ A, const bf16* __restrict__ Bt,
            const float* __restrict__ bias, float* __restrict__ C,
            int R, int K, int N, int relu)
{
  int wave = threadIdx.x >> 5;
  int lane = threadIdx.x & 31;
  int hf   = lane >> 4;     // lane half
  int mr   = lane & 15;     // row (A) / col (B,C) within 16-tile
  int rowTile = blockIdx.x * 4 + wave;   // 16 rows per wave
  int colBase = blockIdx.y * 64;         // 64 cols per block
  long bstride = (long)16 * K;
  const bf16* arow = A  + (long)(rowTile * 16 + mr) * K;   // un-offset (tail)
  const bf16* pa   = arow + hf * 16;                       // fast-path base
  const bf16* brow = Bt + (long)(colBase + mr) * K;        // tile 0, un-offset
  const bf16* pb   = brow + hf * 16;                       // fast-path base
  v8f acc0 = {}, acc1 = {}, acc2 = {}, acc3 = {};
  int K32 = K & ~31;
  for (int kb = 0; kb < K32; kb += 32) {
    v16bf av  = ld_afrag(pa + kb);
    v16bf bv0 = ld_bfrag(pb + kb);
    v16bf bv1 = ld_bfrag(pb + bstride     + kb);
    v16bf bv2 = ld_bfrag(pb + 2 * bstride + kb);
    v16bf bv3 = ld_bfrag(pb + 3 * bstride + kb);
    acc0 = __builtin_amdgcn_wmma_f32_16x16x32_bf16(false, av, false, bv0, (short)0, acc0, false, false);
    acc1 = __builtin_amdgcn_wmma_f32_16x16x32_bf16(false, av, false, bv1, (short)0, acc1, false, false);
    acc2 = __builtin_amdgcn_wmma_f32_16x16x32_bf16(false, av, false, bv2, (short)0, acc2, false, false);
    acc3 = __builtin_amdgcn_wmma_f32_16x16x32_bf16(false, av, false, bv3, (short)0, acc3, false, false);
  }
  if (K32 < K) {            // guarded tail (only K=3 layer hits this)
    int kb = K32;
    v16bf av, bv0, bv1, bv2, bv3;
#pragma unroll
    for (int s = 0; s < 8; ++s) {
      int ka  = kb + hf * 16 + 2 * s;
      int kbi = kb + hf * 16 + 2 * s;
      av[2*s  ] = (ka    < K) ? arow[ka]     : bf0();
      av[2*s+1] = (ka+1  < K) ? arow[ka + 1] : bf0();
      bv0[2*s  ] = (kbi   < K) ? brow[kbi]                   : bf0();
      bv0[2*s+1] = (kbi+1 < K) ? brow[kbi + 1]               : bf0();
      bv1[2*s  ] = (kbi   < K) ? brow[bstride + kbi]         : bf0();
      bv1[2*s+1] = (kbi+1 < K) ? brow[bstride + kbi + 1]     : bf0();
      bv2[2*s  ] = (kbi   < K) ? brow[2 * bstride + kbi]     : bf0();
      bv2[2*s+1] = (kbi+1 < K) ? brow[2 * bstride + kbi + 1] : bf0();
      bv3[2*s  ] = (kbi   < K) ? brow[3 * bstride + kbi]     : bf0();
      bv3[2*s+1] = (kbi+1 < K) ? brow[3 * bstride + kbi + 1] : bf0();
    }
    acc0 = __builtin_amdgcn_wmma_f32_16x16x32_bf16(false, av, false, bv0, (short)0, acc0, false, false);
    acc1 = __builtin_amdgcn_wmma_f32_16x16x32_bf16(false, av, false, bv1, (short)0, acc1, false, false);
    acc2 = __builtin_amdgcn_wmma_f32_16x16x32_bf16(false, av, false, bv2, (short)0, acc2, false, false);
    acc3 = __builtin_amdgcn_wmma_f32_16x16x32_bf16(false, av, false, bv3, (short)0, acc3, false, false);
  }
  v8f accs[4] = {acc0, acc1, acc2, acc3};
#pragma unroll
  for (int t = 0; t < 4; ++t) {
    int ccol = colBase + t * 16 + mr;
    float bs = bias ? bias[ccol] : 0.f;
#pragma unroll
    for (int vr = 0; vr < 8; ++vr) {
      int rr = rowTile * 16 + vr + 8 * hf;  // C: vgpr vr holds M = vr + 8*half
      float val = accs[t][vr] + bs;
      if (relu) val = fmaxf(val, 0.f);
      C[(long)rr * N + ccol] = val;
    }
  }
}

// ---------------------------------------------------------------------------
// Weight conversion
// ---------------------------------------------------------------------------
__global__ void k_wT(const float* __restrict__ W, bf16* __restrict__ Bt,
                     int K, int D, long total) {     // W[K,D] -> Bt[D,K]
  long i = (long)blockIdx.x * blockDim.x + threadIdx.x;
  if (i >= total) return;
  int k = (int)(i % K); int d = (int)(i / K);
  Bt[i] = f2bf(W[(long)k * D + d]);
}
__global__ void k_wdirect(const float* __restrict__ W, bf16* __restrict__ Bt,
                          long total) {              // W[O,C] -> Bt (same layout)
  long i = (long)blockIdx.x * blockDim.x + threadIdx.x;
  if (i >= total) return;
  Bt[i] = f2bf(W[i]);
}
__global__ void k_maskf(const float* __restrict__ mask, float* __restrict__ out,
                        long total) {
  long i = (long)blockIdx.x * blockDim.x + threadIdx.x;
  if (i >= total) return;
  out[i] = tanhf(mask[i]) + 1.f;
}

// ---------------------------------------------------------------------------
// Data BN (feature f over N,T of x transposed to (N, M*V*C, T))
// ---------------------------------------------------------------------------
__global__ void k_dbn_stats(const float* __restrict__ x, float* __restrict__ mean,
                            float* __restrict__ rstd, int Nn, int C, int T, int V, int M) {
  int f = blockIdx.x;
  int m = f / (V * C); int v = (f / C) % V; int c = f % C;
  __shared__ float s1[256], s2[256];
  float a = 0.f, b = 0.f;
  int cnt = Nn * T;
  for (int j = threadIdx.x; j < cnt; j += blockDim.x) {
    int n = j / T, t = j % T;
    float val = x[((((long)n * C + c) * T + t) * V + v) * M + m];
    a += val; b += val * val;
  }
  s1[threadIdx.x] = a; s2[threadIdx.x] = b; __syncthreads();
  for (int s = blockDim.x / 2; s > 0; s >>= 1) {
    if ((int)threadIdx.x < s) { s1[threadIdx.x] += s1[threadIdx.x+s]; s2[threadIdx.x] += s2[threadIdx.x+s]; }
    __syncthreads();
  }
  if (threadIdx.x == 0) {
    float mu = s1[0] / cnt; float var = s2[0] / cnt - mu * mu;
    mean[f] = mu; rstd[f] = rsqrtf(var + BN_EPS);
  }
}
__global__ void k_dbn_apply(const float* __restrict__ x, const float* __restrict__ mean,
                            const float* __restrict__ rstd, const float* __restrict__ g,
                            const float* __restrict__ bb, float* __restrict__ out,
                            int Nn, int C, int T, int V, int M, long total) {
  long i = (long)blockIdx.x * blockDim.x + threadIdx.x;
  if (i >= total) return;
  // out layout [(n*M+m), c, t, v]
  int v  = (int)(i % V); long r = i / V;
  int t  = (int)(r % T); r /= T;
  int c  = (int)(r % C); r /= C;
  int nm = (int)r; int m = nm % M; int n = nm / M;
  int f = (m * V + v) * C + c;
  float val = x[((((long)n * C + c) * T + t) * V + v) * M + m];
  out[i] = (val - mean[f]) * rstd[f] * g[f] + bb[f];
}

// ---------------------------------------------------------------------------
// Shift-GCN: gather (shift_in) + mask -> bf16 rows [(nm*T+t)*V+i, C]
// ---------------------------------------------------------------------------
__global__ void k_gcn_gather(const float* __restrict__ x, const float* __restrict__ maskf,
                             bf16* __restrict__ A, int NM, int C, int T, int V, long total) {
  long i = (long)blockIdx.x * blockDim.x + threadIdx.x;
  if (i >= total) return;
  int j  = (int)(i % C); long r = i / C;
  int vv = (int)(r % V); r /= V;
  int t  = (int)(r % T); int nm = (int)(r / T);
  int src = (vv * C + j * (C + 1)) % (C * V);   // (i*c + j + j*c) mod (c*v)
  int vs = src / C, cs = src % C;
  float val = x[(((long)nm * C + cs) * T + t) * V + vs] * maskf[vv * C + j];
  A[i] = f2bf(val);
}

// stats per feature (i,j) of Y viewed as [R, V, D] (reduce over R)
__global__ void k_colstats(const float* __restrict__ Y, float* __restrict__ mean,
                           float* __restrict__ rstd, int R, int V, int D) {
  int f = blockIdx.x * blockDim.x + threadIdx.x;
  if (f >= V * D) return;
  int i = f / D, j = f % D;
  float a = 0.f, b = 0.f;
  for (int r = 0; r < R; ++r) {
    float v = Y[((long)r * V + i) * D + j];
    a += v; b += v * v;
  }
  float mu = a / R;
  mean[f] = mu; rstd[f] = rsqrtf(b / R - mu * mu + BN_EPS);
}
// per-column stats over big R, one block per column
__global__ void k_colstats_blk(const float* __restrict__ Y, float* __restrict__ mean,
                               float* __restrict__ rstd, long R, int D) {
  int j = blockIdx.x;
  __shared__ float s1[256], s2[256];
  float a = 0.f, b = 0.f;
  for (long r = threadIdx.x; r < R; r += blockDim.x) {
    float v = Y[r * D + j]; a += v; b += v * v;
  }
  s1[threadIdx.x] = a; s2[threadIdx.x] = b; __syncthreads();
  for (int s = blockDim.x / 2; s > 0; s >>= 1) {
    if ((int)threadIdx.x < s) { s1[threadIdx.x] += s1[threadIdx.x+s]; s2[threadIdx.x] += s2[threadIdx.x+s]; }
    __syncthreads();
  }
  if (threadIdx.x == 0) {
    float mu = s1[0] / R; float var = s2[0] / R - mu * mu;
    mean[j] = mu; rstd[j] = rsqrtf(var + BN_EPS);
  }
}

// shift_out permutation folded into BN apply + residual + relu -> [nm, oc, T, V]
__global__ void k_sg_finish(const float* __restrict__ Y, const float* __restrict__ m,
                            const float* __restrict__ rs, const float* __restrict__ g,
                            const float* __restrict__ bb, const float* __restrict__ xres,
                            const float* __restrict__ D, const float* __restrict__ dm,
                            const float* __restrict__ dr, const float* __restrict__ dg,
                            const float* __restrict__ db, float* __restrict__ out,
                            int NM, int OC, int T, int V, int use_down, long total) {
  long idx = (long)blockIdx.x * blockDim.x + threadIdx.x;
  if (idx >= total) return;
  int i  = (int)(idx % V); long r = idx / V;
  int t  = (int)(r % T);  r /= T;
  int j  = (int)(r % OC); int nm = (int)(r / OC);
  int k = i * OC + j;
  long Mvd = (long)OC * V;
  long s = (long)i * OC + j - (long)j * OC;    // shift_out source index
  s %= Mvd; if (s < 0) s += Mvd;
  int i2 = (int)(s / OC), j2 = (int)(s % OC);
  float y = Y[(((long)nm * T + t) * V + i2) * OC + j2];
  float val = (y - m[s]) * rs[s] * g[k] + bb[k];
  float res;
  if (use_down) {
    float dv = D[(((long)nm * T + t) * V + i) * OC + j];
    res = (dv - dm[j]) * dr[j] * dg[j] + db[j];
  } else {
    res = xres[(((long)nm * OC + j) * T + t) * V + i];
  }
  out[idx] = fmaxf(val + res, 0.f);
}

// per-channel BN stats of planar [NM, C, T, V]
__global__ void k_chan_stats(const float* __restrict__ x, float* __restrict__ mean,
                             float* __restrict__ rstd, int NM, int C, int T, int V) {
  int c = blockIdx.x;
  __shared__ float s1[256], s2[256];
  long P = (long)T * V;
  long cnt = (long)NM * P;
  float a = 0.f, b = 0.f;
  for (long j = threadIdx.x; j < cnt; j += blockDim.x) {
    long n = j / P, r = j % P;
    float v = x[(n * C + c) * P + r];
    a += v; b += v * v;
  }
  s1[threadIdx.x] = a; s2[threadIdx.x] = b; __syncthreads();
  for (int s = blockDim.x / 2; s > 0; s >>= 1) {
    if ((int)threadIdx.x < s) { s1[threadIdx.x] += s1[threadIdx.x+s]; s2[threadIdx.x] += s2[threadIdx.x+s]; }
    __syncthreads();
  }
  if (threadIdx.x == 0) {
    float mu = s1[0] / cnt; float var = s2[0] / cnt - mu * mu;
    mean[c] = mu; rstd[c] = rsqrtf(var + BN_EPS);
  }
}

// BN-apply + bilinear shift (stride 1) fused into bf16 GEMM-A build:
// A[(nm*T+t)*V+vv, c] = bilinear(bn(x))(t+xin[c], vv+yin[c])
__global__ void k_tcn_shift1(const float* __restrict__ x, const float* __restrict__ m,
                             const float* __restrict__ rs, const float* __restrict__ g,
                             const float* __restrict__ bb, const float* __restrict__ xin,
                             const float* __restrict__ yin, bf16* __restrict__ A,
                             int NM, int C, int T, int V, long total) {
  long i = (long)blockIdx.x * blockDim.x + threadIdx.x;
  if (i >= total) return;
  int c  = (int)(i % C); long r = i / C;
  int vv = (int)(r % V); r /= V;
  int t  = (int)(r % T); int nm = (int)(r / T);
  float st = (float)t + xin[c];
  float sv = (float)vv + yin[c];
  float t0f = floorf(st), v0f = floorf(sv);
  float wt = st - t0f, wv = sv - v0f;
  int t0 = (int)t0f, v0 = (int)v0f;
  const float* base = x + ((long)nm * C + c) * T * V;
  float mc = m[c], rc = rs[c], gc = g[c], bc = bb[c];
  auto tap = [&](int ti, int vi) -> float {
    if (ti < 0 || ti >= T || vi < 0 || vi >= V) return 0.f;
    return (base[(long)ti * V + vi] - mc) * rc * gc + bc;
  };
  float val = (1.f-wt)*(1.f-wv)*tap(t0, v0) + (1.f-wt)*wv*tap(t0, v0+1)
            + wt*(1.f-wv)*tap(t0+1, v0)    + wt*wv*tap(t0+1, v0+1);
  A[i] = f2bf(val);
}

// bilinear shift with stride from rows-layout Yt[(nm*T+t)*V+vv, c] -> planar [nm,c,To,vv]
__global__ void k_tcn_shift2(const float* __restrict__ Yt, const float* __restrict__ xout,
                             const float* __restrict__ yout, float* __restrict__ out,
                             int NM, int C, int T, int V, int stride, long total) {
  int To = T / stride;
  long i = (long)blockIdx.x * blockDim.x + threadIdx.x;
  if (i >= total) return;
  int vv = (int)(i % V); long r = i / V;
  int to = (int)(r % To); r /= To;
  int c  = (int)(r % C); int nm = (int)(r / C);
  float st = (float)(to * stride) + xout[c];
  float sv = (float)vv + yout[c];
  float t0f = floorf(st), v0f = floorf(sv);
  float wt = st - t0f, wv = sv - v0f;
  int t0 = (int)t0f, v0 = (int)v0f;
  auto tap = [&](int ti, int vi) -> float {
    if (ti < 0 || ti >= T || vi < 0 || vi >= V) return 0.f;
    return Yt[(((long)nm * T + ti) * V + vi) * C + c];
  };
  out[i] = (1.f-wt)*(1.f-wv)*tap(t0, v0) + (1.f-wt)*wv*tap(t0, v0+1)
         + wt*(1.f-wv)*tap(t0+1, v0)    + wt*wv*tap(t0+1, v0+1);
}

// planar [NM,C,T,V] (with t stride) -> bf16 rows [(nm*To+to)*V+vv, C]
__global__ void k_planar_rows(const float* __restrict__ x, bf16* __restrict__ A,
                              int NM, int C, int T, int V, int stride, long total) {
  int To = T / stride;
  long i = (long)blockIdx.x * blockDim.x + threadIdx.x;
  if (i >= total) return;
  int c  = (int)(i % C); long r = i / C;
  int vv = (int)(r % V); r /= V;
  int to = (int)(r % To); int nm = (int)(r / To);
  A[i] = f2bf(x[(((long)nm * C + c) * T + to * stride) * V + vv]);
}

// final BN apply + residual + relu -> planar [nm, c, To, V]
__global__ void k_tcn_finish(const float* __restrict__ o2, const float* __restrict__ m2,
                             const float* __restrict__ r2, const float* __restrict__ g2,
                             const float* __restrict__ b2, const float* __restrict__ xres,
                             const float* __restrict__ D2, const float* __restrict__ rm,
                             const float* __restrict__ rr, const float* __restrict__ rg,
                             const float* __restrict__ rb, float* __restrict__ out,
                             int NM, int C, int To, int V, int kind, long total) {
  long i = (long)blockIdx.x * blockDim.x + threadIdx.x;
  if (i >= total) return;
  int vv = (int)(i % V); long r = i / V;
  int to = (int)(r % To); r /= To;
  int c  = (int)(r % C); int nm = (int)(r / C);
  float v = (o2[i] - m2[c]) * r2[c] * g2[c] + b2[c];
  if (kind == 1) {
    v += xres[i];
  } else if (kind == 2) {
    float d = D2[(((long)nm * To + to) * V + vv) * C + c];
    v += (d - rm[c]) * rr[c] * rg[c] + rb[c];
  }
  out[i] = fmaxf(v, 0.f);
}

// ---------------------------------------------------------------------------
// Head: pooling, FC, argmax, CAM
// ---------------------------------------------------------------------------
__global__ void k_pool(const float* __restrict__ h, float* __restrict__ pooled,
                       int Nn, int M, int C, int T, int V) {
  int i = blockIdx.x * blockDim.x + threadIdx.x;
  if (i >= Nn * C) return;
  int c = i % C, n = i / C;
  long P = (long)T * V;
  float a = 0.f;
  for (int m = 0; m < M; ++m) {
    const float* base = h + ((long)(n * M + m) * C + c) * P;
    for (long r = 0; r < P; ++r) a += base[r];
  }
  pooled[i] = a / (float)(M * T * V);
}
__global__ void k_fc(const float* __restrict__ pooled, const float* __restrict__ W,
                     const float* __restrict__ b, float* __restrict__ logits,
                     int Nn, int K, int C) {
  int i = blockIdx.x * blockDim.x + threadIdx.x;
  if (i >= Nn * K) return;
  int k = i % K, n = i / K;
  float a = b[k];
  for (int c = 0; c < C; ++c) a += pooled[(long)n * C + c] * W[(long)k * C + c];
  logits[i] = a;
}
__global__ void k_argmax(const float* __restrict__ logits, int* __restrict__ pred,
                         int Nn, int K) {
  int n = blockIdx.x * blockDim.x + threadIdx.x;
  if (n >= Nn) return;
  const float* l = logits + (long)n * K;
  int best = 0; float bv = l[0];
  for (int k = 1; k < K; ++k) if (l[k] > bv) { bv = l[k]; best = k; }
  pred[n] = best;
}
__global__ void k_cam(const float* __restrict__ h, const float* __restrict__ W,
                      const int* __restrict__ pred, float* __restrict__ cam,
                      int Nn, int M, int C, int T, int V) {
  int i = blockIdx.x * blockDim.x + threadIdx.x;
  if (i >= Nn * T * V) return;
  int vv = i % V; int t = (i / V) % T; int n = i / (T * V);
  const float* w = W + (long)pred[n] * C;
  long P = (long)T * V;
  float a = 0.f;
  for (int m = 0; m < M; ++m) {
    const float* base = h + (long)(n * M + m) * C * P + (long)t * V + vv;
    for (int c = 0; c < C; ++c) a += base[(long)c * P] * w[c];
  }
  cam[i] = a / (float)M;
}
__global__ void k_camnorm(const float* __restrict__ cam, float* __restrict__ out, int TV) {
  int n = blockIdx.x;
  __shared__ float smin[128], smax[128];
  float lo = 3.4e38f, hi = -3.4e38f;
  for (int i = threadIdx.x; i < TV; i += blockDim.x) {
    float v = cam[(long)n * TV + i];
    lo = fminf(lo, v); hi = fmaxf(hi, v);
  }
  smin[threadIdx.x] = lo; smax[threadIdx.x] = hi; __syncthreads();
  for (int s = blockDim.x / 2; s > 0; s >>= 1) {
    if ((int)threadIdx.x < s) {
      smin[threadIdx.x] = fminf(smin[threadIdx.x], smin[threadIdx.x+s]);
      smax[threadIdx.x] = fmaxf(smax[threadIdx.x], smax[threadIdx.x+s]);
    }
    __syncthreads();
  }
  float mn = smin[0], mx = smax[0];
  float inv = 1.f / (mx - mn);
  for (int i = threadIdx.x; i < TV; i += blockDim.x)
    out[(long)n * TV + i] = (cam[(long)n * TV + i] - mn) * inv;
}

// ---------------------------------------------------------------------------
// Host orchestration
// ---------------------------------------------------------------------------
struct LP {
  const float *mask,*lin_w,*lin_b,*bn_g,*bn_b,*tbn1_g,*tbn1_b,*xin,*yin,
              *tl_w,*tl_b,*xout,*yout,*tbn2_g,*tbn2_b;
  const float *down_w,*down_b,*down_bn_g,*down_bn_b;
  const float *res_w,*res_b,*res_bn_g,*res_bn_b;
};

extern "C" void kernel_launch(void* const* d_in, const int* in_sizes, int n_in,
                              void* d_out, int out_size, void* d_ws, size_t ws_size,
                              hipStream_t stream) {
  (void)in_sizes; (void)n_in; (void)out_size; (void)ws_size;
  const int Nn = 64, C0 = 3, T0 = 64, V = 25, Mp = 2, NM = 128, NCLS = 60, CF = 256;
  const int icfg[10] = {3,64,64,64,64,128,128,128,256,256};
  const int ocfg[10] = {64,64,64,64,128,128,128,256,256,256};
  const int scfg[10] = {1,1,1,1,2,1,1,2,1,1};
  const int rcfg[10] = {0,1,1,1,2,1,1,2,1,1};  // 0 none, 1 id, 2 conv

  // ---- input pointer walking (dict insertion order) ----
  int ii = 0;
  const float* X     = (const float*)d_in[ii++];
  const float* dbn_g = (const float*)d_in[ii++];
  const float* dbn_b = (const float*)d_in[ii++];
  LP L[10];
  for (int l = 0; l < 10; ++l) {
    LP& p = L[l];
    p.mask   = (const float*)d_in[ii++]; p.lin_w = (const float*)d_in[ii++];
    p.lin_b  = (const float*)d_in[ii++]; p.bn_g  = (const float*)d_in[ii++];
    p.bn_b   = (const float*)d_in[ii++]; p.tbn1_g= (const float*)d_in[ii++];
    p.tbn1_b = (const float*)d_in[ii++]; p.xin   = (const float*)d_in[ii++];
    p.yin    = (const float*)d_in[ii++]; p.tl_w  = (const float*)d_in[ii++];
    p.tl_b   = (const float*)d_in[ii++]; p.xout  = (const float*)d_in[ii++];
    p.yout   = (const float*)d_in[ii++]; p.tbn2_g= (const float*)d_in[ii++];
    p.tbn2_b = (const float*)d_in[ii++];
    p.down_w = p.down_b = p.down_bn_g = p.down_bn_b = nullptr;
    p.res_w = p.res_b = p.res_bn_g = p.res_bn_b = nullptr;
    if (icfg[l] != ocfg[l]) {
      p.down_w   = (const float*)d_in[ii++]; p.down_b   = (const float*)d_in[ii++];
      p.down_bn_g= (const float*)d_in[ii++]; p.down_bn_b= (const float*)d_in[ii++];
    }
    if (rcfg[l] == 2) {
      p.res_w   = (const float*)d_in[ii++]; p.res_b   = (const float*)d_in[ii++];
      p.res_bn_g= (const float*)d_in[ii++]; p.res_bn_b= (const float*)d_in[ii++];
    }
  }
  const float* fc_w = (const float*)d_in[ii++];
  const float* fc_b = (const float*)d_in[ii++];

  // ---- workspace partition ----
  const long ACT  = 13107200L;   // max activation (floats)
  const long GMAX = 26214400L;   // max GEMM output (floats)
  char* w = (char*)d_ws;
  size_t off = 0;
  auto take = [&](size_t bytes) -> void* {
    void* p = w + off; off += (bytes + 255) & ~(size_t)255; return p;
  };
  float* actX = (float*)take(ACT * 4);
  float* actG = (float*)take(ACT * 4);
  float* actO = (float*)take(ACT * 4);
  bf16*  Abf  = (bf16*) take(GMAX * 2);
  float* Ybuf = (float*)take(GMAX * 4);
  float* Y2   = (float*)take(GMAX * 4);
  float* sgM  = (float*)take(6400 * 4);
  float* sgR  = (float*)take(6400 * 4);
  float* cM   = (float*)take(256 * 4);
  float* cR   = (float*)take(256 * 4);
  float* dM   = (float*)take(256 * 4);
  float* dR   = (float*)take(256 * 4);
  float* m2   = (float*)take(256 * 4);
  float* r2   = (float*)take(256 * 4);
  float* rM   = (float*)take(256 * 4);
  float* rR   = (float*)take(256 * 4);
  float* dbnM = (float*)take(150 * 4);
  float* dbnR = (float*)take(150 * 4);
  float* mskf = (float*)take(6400 * 4);
  bf16* Wlin  = (bf16*)take(65536 * 2);
  bf16* Wtl   = (bf16*)take(65536 * 2);
  bf16* Wdn   = (bf16*)take(65536 * 2);
  bf16* Wrs   = (bf16*)take(65536 * 2);
  int*  pred  = (int*)  take(64 * 4);
  float* camb = (float*)take(25600 * 4);

  auto g1 = [](long t) { return dim3((unsigned)((t + 255) / 256)); };

  // ---- data BN + reshape to [NM, 3, 64, 25] ----
  k_dbn_stats<<<dim3(Mp * V * C0), dim3(256), 0, stream>>>(X, dbnM, dbnR, Nn, C0, T0, V, Mp);
  long tot0 = (long)NM * C0 * T0 * V;
  k_dbn_apply<<<g1(tot0), dim3(256), 0, stream>>>(X, dbnM, dbnR, dbn_g, dbn_b, actX,
                                                  Nn, C0, T0, V, Mp, tot0);

  // ---- layers ----
  float* cur = actX; float* nxt = actO;
  int Tc = T0;
  for (int l = 0; l < 10; ++l) {
    const LP& P = L[l];
    int ic = icfg[l], oc = ocfg[l], st = scfg[l], rk = rcfg[l];
    int To = Tc / st;
    long rows = (long)NM * Tc * V;
    long rowsTo = (long)NM * To * V;
    int use_down = (ic != oc) ? 1 : 0;

    // ===== shift_gcn =====
    k_maskf<<<g1((long)V * ic), dim3(256), 0, stream>>>(P.mask, mskf, (long)V * ic);
    k_wT<<<g1((long)ic * oc), dim3(256), 0, stream>>>(P.lin_w, Wlin, ic, oc, (long)ic * oc);
    k_gcn_gather<<<g1(rows * ic), dim3(256), 0, stream>>>(cur, mskf, Abf, NM, ic, Tc, V, rows * ic);
    k_gemm_bf16<<<dim3((unsigned)(rows / 64), (unsigned)(oc / 64)), dim3(128), 0, stream>>>(
        Abf, Wlin, nullptr, Ybuf, (int)rows, ic, oc, 0);
    k_colstats<<<g1((long)V * oc), dim3(256), 0, stream>>>(Ybuf, sgM, sgR, NM * Tc, V, oc);
    if (use_down) {
      k_wdirect<<<g1((long)oc * ic), dim3(256), 0, stream>>>(P.down_w, Wdn, (long)oc * ic);
      k_planar_rows<<<g1(rows * ic), dim3(256), 0, stream>>>(cur, Abf, NM, ic, Tc, V, 1, rows * ic);
      k_gemm_bf16<<<dim3((unsigned)(rows / 64), (unsigned)(oc / 64)), dim3(128), 0, stream>>>(
          Abf, Wdn, nullptr, Y2, (int)rows, ic, oc, 0);
      k_colstats_blk<<<dim3(oc), dim3(256), 0, stream>>>(Y2, dM, dR, rows, oc);
    }
    long totg = (long)NM * oc * Tc * V;
    k_sg_finish<<<g1(totg), dim3(256), 0, stream>>>(
        Ybuf, sgM, sgR, P.bn_g, P.bn_b, cur, Y2, dM, dR, P.down_bn_g, P.down_bn_b,
        actG, NM, oc, Tc, V, use_down, totg);

    // ===== shift_tcn =====
    k_chan_stats<<<dim3(oc), dim3(256), 0, stream>>>(actG, cM, cR, NM, oc, Tc, V);
    k_wdirect<<<g1((long)oc * oc), dim3(256), 0, stream>>>(P.tl_w, Wtl, (long)oc * oc);
    k_tcn_shift1<<<g1(rows * oc), dim3(256), 0, stream>>>(
        actG, cM, cR, P.tbn1_g, P.tbn1_b, P.xin, P.yin, Abf, NM, oc, Tc, V, rows * oc);
    k_gemm_bf16<<<dim3((unsigned)(rows / 64), (unsigned)(oc / 64)), dim3(128), 0, stream>>>(
        Abf, Wtl, P.tl_b, Y2, (int)rows, oc, oc, 1);
    long tot2 = (long)NM * oc * To * V;
    k_tcn_shift2<<<g1(tot2), dim3(256), 0, stream>>>(Y2, P.xout, P.yout, actG,
                                                     NM, oc, Tc, V, st, tot2);
    k_chan_stats<<<dim3(oc), dim3(256), 0, stream>>>(actG, m2, r2, NM, oc, To, V);
    if (rk == 2) {
      k_wdirect<<<g1((long)oc * ic), dim3(256), 0, stream>>>(P.res_w, Wrs, (long)oc * ic);
      k_planar_rows<<<g1(rowsTo * ic), dim3(256), 0, stream>>>(cur, Abf, NM, ic, Tc, V, st, rowsTo * ic);
      k_gemm_bf16<<<dim3((unsigned)(rowsTo / 64), (unsigned)(oc / 64)), dim3(128), 0, stream>>>(
          Abf, Wrs, nullptr, Ybuf, (int)rowsTo, ic, oc, 0);
      k_colstats_blk<<<dim3(oc), dim3(256), 0, stream>>>(Ybuf, rM, rR, rowsTo, oc);
    }
    k_tcn_finish<<<g1(tot2), dim3(256), 0, stream>>>(
        actG, m2, r2, P.tbn2_g, P.tbn2_b, cur, Ybuf, rM, rR, P.res_bn_g, P.res_bn_b,
        nxt, NM, oc, To, V, rk, tot2);

    float* tmp = cur; cur = nxt; nxt = tmp;
    Tc = To;
  }

  // ---- head: pooled / logits / argmax / CAM ----
  float* outF   = (float*)d_out;
  float* logits = outF;                      // [64, 60]
  float* pooled = outF + (long)Nn * NCLS;    // [64, 256]
  float* cams   = pooled + (long)Nn * CF;    // [64, 16, 25]
  const int Tf = 16;
  k_pool<<<g1((long)Nn * CF), dim3(256), 0, stream>>>(cur, pooled, Nn, Mp, CF, Tf, V);
  k_fc<<<g1((long)Nn * NCLS), dim3(256), 0, stream>>>(pooled, fc_w, fc_b, logits, Nn, NCLS, CF);
  k_argmax<<<dim3(1), dim3(64), 0, stream>>>(logits, pred, Nn, NCLS);
  k_cam<<<g1((long)Nn * Tf * V), dim3(256), 0, stream>>>(cur, fc_w, pred, camb, Nn, Mp, CF, Tf, V);
  k_camnorm<<<dim3(Nn), dim3(128), 0, stream>>>(camb, cams, Tf * V);
}
